// DynamicContextModule3d_25890062861005
// MI455X (gfx1250) — matrix operand order repaired and checked by reading, not compiled, not run on hardware
//
#include <hip/hip_runtime.h>

// ---------------- problem constants ----------------
#define NB     8
#define CH     64
#define DIM    48
#define SP     (48*48*48)      // 110592 spatial per channel
// main-kernel tiling
#define TD     4
#define TTH    4
#define TW     8
#define NPOS   (TD*TTH*TW)     // 128 outputs per WG
#define HD     6
#define HHH    6
#define HW     10
#define HALO   (HD*HHH*HW)     // 360 halo positions
#define HALOP  368             // padded to multiple of 16
#define XSTR   66              // channel-dim stride for xs/cs/weight LDS (bank-friendly)
#define USTR   376             // spatial stride for up LDS

typedef __attribute__((ext_vector_type(16))) __bf16 v16bf;
typedef __attribute__((ext_vector_type(8)))  float  v8f;
typedef __attribute__((ext_vector_type(4)))  unsigned int u32x4;
typedef __attribute__((ext_vector_type(8)))  int          i32x8;
typedef __attribute__((ext_vector_type(4)))  int          i32x4;

union ABfrag { unsigned int u[8]; v16bf v; };

__device__ __forceinline__ unsigned short f2bf(float f) {
    union { float f; unsigned int u; } x; x.f = f;
    unsigned int u = x.u;
    unsigned int r = (u + 0x7FFFu + ((u >> 16) & 1u)) >> 16;   // RNE
    return (unsigned short)r;
}
__device__ __forceinline__ float bf2f(unsigned short b) {
    union { unsigned int u; float f; } x; x.u = ((unsigned int)b) << 16;
    return x.f;
}

__device__ __forceinline__ v8f wmma_bf16(const ABfrag& a, const ABfrag& b, v8f c) {
    return __builtin_amdgcn_wmma_f32_16x16x32_bf16(
        /*neg_a=*/false, a.v, /*neg_b=*/false, b.v,
        /*c_mod=*/(short)0, c, /*reuse_a=*/false, /*reuse_b=*/false);
}

// ============================================================
// Kernel 1: block-mean pool (16^3) + dyn = w_down @ pooled + b_down
// grid = NB*27, block = 256
// ============================================================
__global__ __launch_bounds__(256)
void k_pool_dyn(const float* __restrict__ x,
                const float* __restrict__ w_down,
                const float* __restrict__ b_down,
                float* __restrict__ dyn) {
    __shared__ float pooledL[CH];
    __shared__ float wsum[8];

    const int bi  = blockIdx.x;
    const int n   = bi / 27;
    const int blk = bi % 27;
    const int iz = blk / 9, iy = (blk / 3) % 3, ix = blk % 3;

    const int tid  = threadIdx.x;
    const int lane = tid & 31;
    const int wav  = tid >> 5;
    const int dd = tid >> 4, hh = tid & 15;   // (d,h) row inside the 16^3 block

    for (int c = 0; c < CH; ++c) {
        const float* p = x + ((((size_t)(n*CH + c))*DIM + (iz*16 + dd))*DIM + (iy*16 + hh))*DIM + ix*16;
        float s = 0.f;
        #pragma unroll
        for (int w = 0; w < 16; ++w) s += p[w];
        #pragma unroll
        for (int off = 16; off; off >>= 1) s += __shfl_xor(s, off, 32);
        if (lane == 0) wsum[wav] = s;
        __syncthreads();
        if (tid == 0) {
            float t = 0.f;
            #pragma unroll
            for (int q = 0; q < 8; ++q) t += wsum[q];
            pooledL[c] = t * (1.f / 4096.f);
        }
        __syncthreads();
    }
    if (tid < CH) {
        float acc = b_down[tid];
        for (int c = 0; c < CH; ++c) acc += w_down[tid*CH + c] * pooledL[c];
        dyn[((size_t)n*CH + tid)*27 + blk] = acc;
    }
}

// ============================================================
// Kernel 2: fused  up-GEMM (WMMA) -> dynamic depthwise 3^3 -> final-GEMM (WMMA)
// grid = NB * 12*12*6, block = 256 (8 waves)
// ============================================================
__global__ __launch_bounds__(256)
void k_main(const float* __restrict__ x,
            const float* __restrict__ w_up,   const float* __restrict__ b_up,
            const float* __restrict__ w_fin,  const float* __restrict__ b_fin,
            const float* __restrict__ dyn,
            float* __restrict__ out) {
    __shared__ unsigned short xs [HALOP * XSTR]; // x tile bf16, [spatial][channel]; reused as conv result [pos][channel]
    __shared__ unsigned short upl[CH * USTR];    // up tile bf16, [channel][spatial]
    __shared__ unsigned short wuL[CH * XSTR];    // w_up  bf16 [o][c]
    __shared__ unsigned short wfL[CH * XSTR];    // w_fin bf16 [o][c]
    __shared__ float dynl[CH * 27];
    __shared__ float bupL[CH];
    __shared__ float bfnL[CH];

    const int tid  = threadIdx.x;
    const int lane = tid & 31;
    const int wav  = tid >> 5;
    const int row  = lane & 15;
    const int hv   = lane >> 4;

    int b = blockIdx.x;
    const int n  = b / (12*12*6); b %= (12*12*6);
    const int bd = b / (12*6);    b %= (12*6);
    const int bh = b / 6;
    const int bw = b % 6;
    const int D0 = bd*TD, H0 = bh*TTH, W0 = bw*TW;

    // ---------------- phase 0: stage to LDS ----------------
    // dyn taps: raw f32, contiguous, no conversion -> Tensor Data Mover (wave 0 issues)
    if (wav == 0) {
        const unsigned lds_base = (unsigned)(size_t)(void*)&dynl[0];   // flat LDS addr: offset in [31:0]
        const unsigned long long ga = (unsigned long long)(size_t)(dyn + (size_t)n*CH*27);
        const unsigned nelem = CH*27;                                  // 1728 dwords
        u32x4 g0;
        g0[0] = 1u;                                                    // count=1, user mode
        g0[1] = lds_base;                                              // lds_addr
        g0[2] = (unsigned)(ga & 0xFFFFFFFFu);                          // global_addr[31:0]
        g0[3] = (unsigned)((ga >> 32) & 0x01FFFFFFu) | (2u << 30);     // global_addr[56:32] | type=2
        i32x8 g1;
        g1[0] = (int)(2u << 16);                                       // data_size=4B, no multicast
        g1[1] = (int)((nelem & 0xFFFFu) << 16);                        // tensor_dim0[15:0]
        g1[2] = (int)((nelem >> 16) | (1u << 16));                     // tensor_dim0[31:16] | tensor_dim1=1
        g1[3] = (int)(nelem << 16);                                    // tile_dim0 = nelem
        g1[4] = 1;                                                     // tile_dim1=1, tile_dim2=0
        g1[5] = (int)nelem;                                            // tensor_dim0_stride
        g1[6] = 0; g1[7] = 0;
        i32x4 gz = {0, 0, 0, 0};
        i32x8 gz8 = {0, 0, 0, 0, 0, 0, 0, 0};
        __builtin_amdgcn_tensor_load_to_lds(g0, g1, gz, gz, gz8, 0);
        __builtin_amdgcn_s_wait_tensorcnt(0);
    }

    for (int i = tid; i < CH*CH; i += 256) {
        const int o = i >> 6, c = i & 63;
        wuL[o*XSTR + c] = f2bf(w_up[i]);
        wfL[o*XSTR + c] = f2bf(w_fin[i]);
    }
    if (tid < CH) { bupL[tid] = b_up[tid]; bfnL[tid] = b_fin[tid]; }

    // x tile (with halo, zero-padded outside the domain), bf16 transposed
    for (int e = tid; e < HALO*CH; e += 256) {
        const int c = e / HALO;
        const int s = e % HALO;
        const int wz = s % HW;
        const int rs = s / HW;
        const int hy = rs % HHH;
        const int dz = rs / HHH;
        const int gd = D0 - 1 + dz, gh = H0 - 1 + hy, gw = W0 - 1 + wz;
        float v = 0.f;
        if (gd >= 0 && gd < DIM && gh >= 0 && gh < DIM && gw >= 0 && gw < DIM)
            v = x[((((size_t)(n*CH + c))*DIM + gd)*DIM + gh)*DIM + gw];
        xs[s*XSTR + c] = f2bf(v);
    }
    for (int i = tid; i < (HALOP - HALO)*XSTR; i += 256) xs[HALO*XSTR + i] = 0;
    __syncthreads();

    // ---------------- phase 1: up = w_up @ x + b_up over halo (WMMA) ----------------
    {
        const int nNT = HALOP / 16;            // 23 spatial tiles
        for (int t = wav; t < 4*nNT; t += 8) {
            const int mt = t / nNT, nt = t % nNT;
            const int m0 = mt*16, s0 = nt*16;
            ABfrag a0, a1, bb0, bb1;
            #pragma unroll
            for (int v = 0; v < 8; ++v) {
                const int kk = ((v & 3)*2) + ((v >> 2)*16) + hv*8;  // A: 16-bit 16x32 layout
                a0.u[v] = *(const unsigned int*)&wuL[(m0 + row)*XSTR + kk];
                a1.u[v] = *(const unsigned int*)&wuL[(m0 + row)*XSTR + 32 + kk];
                const int kb = ((v & 3)*2) + ((v >> 2)*16) + hv*8;  // B: same k order as A
                bb0.u[v] = *(const unsigned int*)&xs[(s0 + row)*XSTR + kb];
                bb1.u[v] = *(const unsigned int*)&xs[(s0 + row)*XSTR + 32 + kb];
            }
            v8f c = {};
            c = wmma_bf16(a0, bb0, c);
            c = wmma_bf16(a1, bb1, c);
            const int s = s0 + row;
            #pragma unroll
            for (int r = 0; r < 8; ++r) {
                const int ch = m0 + r + hv*8;
                upl[ch*USTR + s] = f2bf(c[r] + bupL[ch]);
            }
        }
    }
    __syncthreads();

    // ---------------- phase 2: dynamic depthwise 3x3x3 ----------------
    {
        // per-block w masks (uniform) folded into the row values
        float wmask[HW];
        #pragma unroll
        for (int q = 0; q < HW; ++q) {
            const int gw = W0 - 1 + q;
            wmask[q] = (gw >= 0 && gw < DIM) ? 1.f : 0.f;
        }
        for (int task = tid; task < CH*TD*TTH; task += 256) {
            const int ch = task & 63;
            const int od = (task >> 6) & 3;
            const int oh = task >> 8;
            float vd[3], vh[3];
            #pragma unroll
            for (int q = 0; q < 3; ++q) {
                const int gd = D0 + od + q - 1;
                const int gh = H0 + oh + q - 1;
                vd[q] = (gd >= 0 && gd < DIM) ? 1.f : 0.f;
                vh[q] = (gh >= 0 && gh < DIM) ? 1.f : 0.f;
            }
            const float* dt = &dynl[ch*27];
            float acc[TW];
            #pragma unroll
            for (int q = 0; q < TW; ++q) acc[q] = 0.f;
            #pragma unroll
            for (int tz = 0; tz < 3; ++tz) {
                #pragma unroll
                for (int ty = 0; ty < 3; ++ty) {
                    const float vdh = vd[tz]*vh[ty];
                    const unsigned short* ub = &upl[ch*USTR + ((od + tz)*HHH + (oh + ty))*HW];
                    float rowv[HW];
                    #pragma unroll
                    for (int q = 0; q < HW; ++q) rowv[q] = wmask[q]*bf2f(ub[q]);
                    #pragma unroll
                    for (int tx = 0; tx < 3; ++tx) {
                        const float wt = vdh * dt[(tz*3 + ty)*3 + tx];
                        #pragma unroll
                        for (int ow = 0; ow < TW; ++ow) acc[ow] += wt * rowv[ow + tx];
                    }
                }
            }
            const int pbase = (od*TTH + oh)*TW;
            #pragma unroll
            for (int ow = 0; ow < TW; ++ow)
                xs[(pbase + ow)*XSTR + ch] = f2bf(acc[ow]);   // reuse xs as conv staging [pos][ch]
        }
    }
    __syncthreads();

    // ---------------- phase 3: out = w_final @ conv + b_final (WMMA) ----------------
    {
        const size_t obase0 = (size_t)n*CH*SP;
        for (int t = wav; t < 4*(NPOS/16); t += 8) {   // 4 M-tiles x 8 N-tiles
            const int mt = t >> 3, nt = t & 7;
            const int m0 = mt*16, s0 = nt*16;
            ABfrag a0, a1, bb0, bb1;
            #pragma unroll
            for (int v = 0; v < 8; ++v) {
                const int kk = ((v & 3)*2) + ((v >> 2)*16) + hv*8;
                a0.u[v] = *(const unsigned int*)&wfL[(m0 + row)*XSTR + kk];
                a1.u[v] = *(const unsigned int*)&wfL[(m0 + row)*XSTR + 32 + kk];
                const int kb = ((v & 3)*2) + ((v >> 2)*16) + hv*8;
                bb0.u[v] = *(const unsigned int*)&xs[(s0 + row)*XSTR + kb];
                bb1.u[v] = *(const unsigned int*)&xs[(s0 + row)*XSTR + 32 + kb];
            }
            v8f c = {};
            c = wmma_bf16(a0, bb0, c);
            c = wmma_bf16(a1, bb1, c);
            const int p  = s0 + row;
            const int ow = p & 7, oh = (p >> 3) & 3, od = p >> 5;
            const size_t sbase = obase0 + ((size_t)(D0 + od)*DIM + (H0 + oh))*DIM + (W0 + ow);
            #pragma unroll
            for (int r = 0; r < 8; ++r) {
                const int o = m0 + r + hv*8;
                out[sbase + (size_t)o*SP] = c[r] + bfnL[o];
            }
        }
    }
}

extern "C" void kernel_launch(void* const* d_in, const int* in_sizes, int n_in,
                              void* d_out, int out_size, void* d_ws, size_t ws_size,
                              hipStream_t stream) {
    const float* x       = (const float*)d_in[0];
    const float* w_up    = (const float*)d_in[1];
    const float* b_up    = (const float*)d_in[2];
    const float* w_down  = (const float*)d_in[3];
    const float* b_down  = (const float*)d_in[4];
    const float* w_final = (const float*)d_in[5];
    const float* b_final = (const float*)d_in[6];
    float* out = (float*)d_out;
    float* dyn = (float*)d_ws;                    // NB*CH*27 floats = 55 KB

    k_pool_dyn<<<NB*27, 256, 0, stream>>>(x, w_down, b_down, dyn);
    k_main<<<NB*12*12*6, 256, 0, stream>>>(x, w_up, b_up, w_final, b_final, dyn, out);
}
